// ViTEncoderLayer_46677704573477
// MI455X (gfx1250) — compile-verified, matrix operand order checked by reading, not run on hardware
//
#include <hip/hip_runtime.h>
#include <math.h>

// ---------------------------------------------------------------------------
// Types for CDNA5 WMMA (wave32): v_wmma_f32_16x16x32_bf16
// ---------------------------------------------------------------------------
typedef __attribute__((ext_vector_type(16))) __bf16 v16bf;
typedef __attribute__((ext_vector_type(8)))  float  v8f;
typedef __attribute__((ext_vector_type(4)))  float  v4f;
typedef __attribute__((ext_vector_type(4)))  int    v4i;

union Frag {
    v16bf v;
    v4f   f4[2];
    unsigned short e[16];
};

// fp32 -> bf16 round-to-nearest-even (bits in ushort)
static __device__ __forceinline__ unsigned short f2bf(float f) {
    unsigned int u = __float_as_uint(f);
    unsigned int r = (u + 0x7FFFu + ((u >> 16) & 1u)) >> 16;
    return (unsigned short)r;
}

// ---------------------------------------------------------------------------
// Async global->LDS copies (GLOBAL_LOAD_ASYNC_TO_LDS_B128, ASYNCcnt-tracked).
// Builtin signature (from compiler diagnostic): (int4 AS1*, int4 AS3*, Ii, Ii).
// Guarded: device pass on gfx1250 with the builtin -> async path; otherwise
// (host pass / older toolchain) plain v4f copies through VGPRs.
// ---------------------------------------------------------------------------
#if defined(__gfx1250__) && __has_builtin(__builtin_amdgcn_global_load_async_to_lds_b128)
#define USE_ASYNC_LDS 1
#else
#define USE_ASYNC_LDS 0
#endif

static __device__ __forceinline__ void copy16_g2l(const unsigned short* g,
                                                  unsigned short* l) {
#if USE_ASYNC_LDS
    __builtin_amdgcn_global_load_async_to_lds_b128(
        (__attribute__((address_space(1))) v4i*)(g),
        (__attribute__((address_space(3))) v4i*)(l), 0, 0);
#else
    *(v4f*)l = *(const v4f*)g;
#endif
}

template <int N>
static __device__ __forceinline__ void wait_async() {
#if USE_ASYNC_LDS
#if __has_builtin(__builtin_amdgcn_s_wait_asynccnt)
    __builtin_amdgcn_s_wait_asynccnt(N);
#else
    asm volatile("s_wait_asynccnt %0" ::"i"(N));
#endif
#endif
}

// Load one 16x32 bf16 A/B fragment (NT layout). Lane l covers row = l&15,
// K-half = l>>4: K {kh*8..kh*8+7} and {kh*8+16..kh*8+23}.
static __device__ __forceinline__ v16bf load_frag_nt(const unsigned short* base,
                                                     int stride, int lane) {
    int row = lane & 15;
    int kh  = (lane >> 4) << 3;
    const unsigned short* p = base + (size_t)row * stride + kh;
    Frag u;
    u.f4[0] = *(const v4f*)(p);
    u.f4[1] = *(const v4f*)(p + 16);
    return u.v;
}

static __device__ __forceinline__ v8f wmma_bf16(v16bf a, v16bf b, v8f c) {
    return __builtin_amdgcn_wmma_f32_16x16x32_bf16(false, a, false, b,
                                                   (short)0, c, false, false);
}

// ---------------------------------------------------------------------------
// fp32 -> bf16 conversion (weights)
// ---------------------------------------------------------------------------
__global__ __launch_bounds__(256) void cvt_bf16_kernel(const float* __restrict__ in,
                                                       unsigned short* __restrict__ out,
                                                       int n) {
    int i = blockIdx.x * 256 + threadIdx.x;
    if (i < n) out[i] = f2bf(in[i]);
}

// ---------------------------------------------------------------------------
// LayerNorm (fp32 in, bf16 out), one 256-thread block per row, D=768
// ---------------------------------------------------------------------------
__global__ __launch_bounds__(256) void layernorm_bf16_kernel(
        const float* __restrict__ x, const float* __restrict__ w,
        const float* __restrict__ bwt, unsigned short* __restrict__ out, int D) {
    const int row = blockIdx.x;
    const int tid = threadIdx.x;
    const float* xr = x + (size_t)row * D;
    float v[3], s = 0.f, s2 = 0.f;
#pragma unroll
    for (int i = 0; i < 3; ++i) {
        v[i] = xr[tid + i * 256];
        s += v[i];
        s2 += v[i] * v[i];
    }
#pragma unroll
    for (int m = 16; m >= 1; m >>= 1) {
        s  += __shfl_xor(s,  m, 32);
        s2 += __shfl_xor(s2, m, 32);
    }
    __shared__ float rs[8], rs2[8];
    int wv = tid >> 5, lane = tid & 31;
    if (lane == 0) { rs[wv] = s; rs2[wv] = s2; }
    __syncthreads();
    float ts = 0.f, ts2 = 0.f;
#pragma unroll
    for (int i = 0; i < 8; ++i) { ts += rs[i]; ts2 += rs2[i]; }
    float mu   = ts / (float)D;
    float var  = ts2 / (float)D - mu * mu;
    float rstd = rsqrtf(var + 1e-12f);
#pragma unroll
    for (int i = 0; i < 3; ++i) {
        int c = tid + i * 256;
        out[(size_t)row * D + c] = f2bf((v[i] - mu) * rstd * w[c] + bwt[c]);
    }
}

// ---------------------------------------------------------------------------
// Generic NT bf16 GEMM:  C[M,N] = A[M,K] * B[N,K]^T  (+ epilogue)
//  EPI 0: store bf16
//  EPI 1: += bias[col] + resid[row,col], store fp32
//  EPI 2: += bias[col], exact GELU, store bf16
// Block: 256 threads = 8 waves; block tile 128x128; wave tile 32x64.
// Double-buffered LDS, async global->LDS staging, software pipelined on
// ASYNCcnt (4 outstanding b128 per thread per tile).
// ---------------------------------------------------------------------------
template <int EPI>
__global__ __launch_bounds__(256) void gemm_nt_bf16_kernel(
        const unsigned short* __restrict__ A, const unsigned short* __restrict__ Bw,
        int M, int N, int K, void* __restrict__ Cout,
        const float* __restrict__ bias, const float* __restrict__ resid) {
    __shared__ alignas(16) unsigned short sA[2][128 * 40];  // 128 x 32, stride 40
    __shared__ alignas(16) unsigned short sB[2][128 * 40];  // 128 x 32, stride 40

    const int tid   = threadIdx.x;
    const int lane  = tid & 31;
    const int wid   = tid >> 5;
    const int waveM = wid >> 1;            // 0..3  (rows of 32)
    const int waveN = wid & 1;             // 0..1  (cols of 64)
    const int mBlk  = blockIdx.y * 128;
    const int nBlk  = blockIdx.x * 128;

    // staging coordinates: each thread moves 16 elems of A and 16 of B
    const int srow = tid >> 1;             // 0..127
    const int sseg = (tid & 1) * 16;       // 0 or 16
    const unsigned short* gA = A  + (size_t)(mBlk + srow) * K + sseg;
    const unsigned short* gB = Bw + (size_t)(nBlk + srow) * K + sseg;
    unsigned short* lA = &sA[0][srow * 40 + sseg];
    unsigned short* lB = &sB[0][srow * 40 + sseg];
    const int bufStride = 128 * 40;

    auto stage = [&](int buf, int k0) {
        copy16_g2l(gA + k0,     lA + buf * bufStride);
        copy16_g2l(gA + k0 + 8, lA + buf * bufStride + 8);
        copy16_g2l(gB + k0,     lB + buf * bufStride);
        copy16_g2l(gB + k0 + 8, lB + buf * bufStride + 8);
    };

    v8f acc[2][4] = {};

    stage(0, 0);                                     // prologue
    for (int k0 = 0; k0 < K; k0 += 32) {
        const int cur = (k0 >> 5) & 1;
        if (k0 + 32 < K) {
            stage(cur ^ 1, k0 + 32);                 // prefetch next tile
            if (k0 + 64 < K) {                       // L2 prefetch 2 tiles ahead
                __builtin_prefetch(gA + k0 + 64, 0, 1);
                __builtin_prefetch(gB + k0 + 64, 0, 1);
            }
            wait_async<4>();                         // drain current tile's 4 ops
        } else {
            wait_async<0>();
        }
        __syncthreads();

        const unsigned short* pa = &sA[cur][(waveM * 32) * 40];
        const unsigned short* pb = &sB[cur][(waveN * 64) * 40];
        v16bf af[2], bfr[4];
        af[0] = load_frag_nt(pa,            40, lane);
        af[1] = load_frag_nt(pa + 16 * 40,  40, lane);
#pragma unroll
        for (int nt = 0; nt < 4; ++nt)
            bfr[nt] = load_frag_nt(pb + nt * 16 * 40, 40, lane);
#pragma unroll
        for (int mt = 0; mt < 2; ++mt)
#pragma unroll
            for (int nt = 0; nt < 4; ++nt)
                acc[mt][nt] = wmma_bf16(af[mt], bfr[nt], acc[mt][nt]);
        __syncthreads();
    }

    // epilogue — C layout: VGPR r -> row r (lanes 0-15) / row 8+r (lanes 16-31)
    const int rHalf = (lane >> 4) * 8;
    const int colL  = lane & 15;
#pragma unroll
    for (int mt = 0; mt < 2; ++mt) {
#pragma unroll
        for (int nt = 0; nt < 4; ++nt) {
            int colBase = nBlk + waveN * 64 + nt * 16 + colL;
            float bv = (EPI != 0) ? bias[colBase] : 0.f;
#pragma unroll
            for (int r = 0; r < 8; ++r) {
                int row = mBlk + waveM * 32 + mt * 16 + rHalf + r;
                float vv = acc[mt][nt][r];
                if (EPI == 1) {
                    vv += bv + resid[(size_t)row * N + colBase];
                    ((float*)Cout)[(size_t)row * N + colBase] = vv;
                } else if (EPI == 2) {
                    vv += bv;
                    vv = 0.5f * vv * (1.0f + erff(vv * 0.70710678118654752f));
                    ((unsigned short*)Cout)[(size_t)row * N + colBase] = f2bf(vv);
                } else {
                    ((unsigned short*)Cout)[(size_t)row * N + colBase] = f2bf(vv);
                }
            }
        }
    }
}

// ---------------------------------------------------------------------------
// Flash attention: per (b,h), 64 query rows per block (4 waves x 16 rows),
// key tiles of 32, DH=64, T=1024, D=768, scale = 1/8.
// ---------------------------------------------------------------------------
__global__ __launch_bounds__(128) void attn_kernel(
        const unsigned short* __restrict__ Q, const unsigned short* __restrict__ Km,
        const unsigned short* __restrict__ V, unsigned short* __restrict__ O) {
    __shared__ alignas(16) unsigned short sK[32 * 72];     // 32 keys x 64 d
    __shared__ alignas(16) unsigned short sVt[64 * 40];    // 64 d x 32 keys
    __shared__ alignas(16) unsigned short sP[4][16 * 40];  // per-wave P tile

    const int tid  = threadIdx.x;
    const int lane = tid & 31;
    const int w    = tid >> 5;
    const int b    = blockIdx.z;
    const int h    = blockIdx.y;
    const int q0   = blockIdx.x * 64 + w * 16;
    const int D = 768, T = 1024, DH = 64;

    const unsigned short* Qbase = Q + ((size_t)(b * T + q0)) * D + h * DH;
    v16bf qf[2];
    qf[0] = load_frag_nt(Qbase,      D, lane);
    qf[1] = load_frag_nt(Qbase + 32, D, lane);

    float mi[8], li[8];
#pragma unroll
    for (int r = 0; r < 8; ++r) { mi[r] = -1e30f; li[r] = 0.f; }
    v8f oacc[4] = {};
    const float scale = 0.125f;

    for (int kt = 0; kt < T; kt += 32) {
        __syncthreads();
        {   // stage K tile via async-to-LDS; V tile transposed via VGPRs
            int krow = tid >> 2;
            int dseg = (tid & 3) * 16;
            const unsigned short* gk = Km + ((size_t)(b * T + kt + krow)) * D + h * DH + dseg;
            copy16_g2l(gk,     &sK[krow * 72 + dseg]);
            copy16_g2l(gk + 8, &sK[krow * 72 + dseg + 8]);

            const unsigned short* gv = V + ((size_t)(b * T + kt + krow)) * D + h * DH + dseg;
            Frag vv;
            vv.f4[0] = *(const v4f*)gv;
            vv.f4[1] = *(const v4f*)(gv + 8);
#pragma unroll
            for (int i = 0; i < 16; ++i)
                sVt[(dseg + i) * 40 + krow] = vv.e[i];
        }
        wait_async<0>();
        __syncthreads();

        // S = Q K^T
        v8f s0 = {}, s1 = {};
#pragma unroll
        for (int c = 0; c < 2; ++c) {
            v16bf kb0 = load_frag_nt(&sK[ 0 * 72 + c * 32], 72, lane);
            v16bf kb1 = load_frag_nt(&sK[16 * 72 + c * 32], 72, lane);
            s0 = wmma_bf16(qf[c], kb0, s0);
            s1 = wmma_bf16(qf[c], kb1, s1);
        }

        // online softmax (rows in 16-lane halves; xor-reduce over N)
        float mnew[8], p0[8], p1[8], corr[8];
#pragma unroll
        for (int r = 0; r < 8; ++r) {
            float mx = fmaxf(s0[r], s1[r]);
            mx = fmaxf(mx, __shfl_xor(mx, 1, 32));
            mx = fmaxf(mx, __shfl_xor(mx, 2, 32));
            mx = fmaxf(mx, __shfl_xor(mx, 4, 32));
            mx = fmaxf(mx, __shfl_xor(mx, 8, 32));
            mnew[r] = fmaxf(mi[r], mx * scale);
        }
#pragma unroll
        for (int r = 0; r < 8; ++r) {
            p0[r] = __expf(s0[r] * scale - mnew[r]);
            p1[r] = __expf(s1[r] * scale - mnew[r]);
            float sum = p0[r] + p1[r];
            sum += __shfl_xor(sum, 1, 32);
            sum += __shfl_xor(sum, 2, 32);
            sum += __shfl_xor(sum, 4, 32);
            sum += __shfl_xor(sum, 8, 32);
            corr[r] = __expf(mi[r] - mnew[r]);
            li[r] = li[r] * corr[r] + sum;
            mi[r] = mnew[r];
        }
#pragma unroll
        for (int dt = 0; dt < 4; ++dt)
#pragma unroll
            for (int r = 0; r < 8; ++r)
                oacc[dt][r] *= corr[r];

        // C-layout -> A-layout via per-wave LDS round trip
        {
            const int rHalf = (lane >> 4) * 8;
            const int colL  = lane & 15;
            unsigned short* pw = &sP[w][0];
#pragma unroll
            for (int r = 0; r < 8; ++r) {
                pw[(rHalf + r) * 40 + colL]      = f2bf(p0[r]);
                pw[(rHalf + r) * 40 + colL + 16] = f2bf(p1[r]);
            }
        }
        v16bf pf = load_frag_nt(&sP[w][0], 40, lane);
#pragma unroll
        for (int dt = 0; dt < 4; ++dt) {
            v16bf vb = load_frag_nt(&sVt[dt * 16 * 40], 40, lane);
            oacc[dt] = wmma_bf16(pf, vb, oacc[dt]);
        }
    }

    const int rHalf = (lane >> 4) * 8;
    const int colL  = lane & 15;
#pragma unroll
    for (int r = 0; r < 8; ++r) {
        float inv = 1.0f / li[r];
        int row = q0 + rHalf + r;
        unsigned short* op = O + ((size_t)(b * T + row)) * D + h * DH + colL;
#pragma unroll
        for (int dt = 0; dt < 4; ++dt)
            op[dt * 16] = f2bf(oacc[dt][r] * inv);
    }
}

// ---------------------------------------------------------------------------
// Launch: LN1 -> QKV GEMMs -> flash attn -> out-proj(+res) -> LN2 ->
//         FFN1(+GELU) -> FFN2(+res) -> d_out
// ---------------------------------------------------------------------------
extern "C" void kernel_launch(void* const* d_in, const int* in_sizes, int n_in,
                              void* d_out, int out_size, void* d_ws, size_t ws_size,
                              hipStream_t stream) {
    (void)in_sizes; (void)n_in; (void)out_size; (void)ws_size;
    const float* x    = (const float*)d_in[0];
    const float* Wq   = (const float*)d_in[1];
    const float* Wk   = (const float*)d_in[2];
    const float* Wv   = (const float*)d_in[3];
    const float* Wo   = (const float*)d_in[4];
    const float* bo   = (const float*)d_in[5];
    const float* W1   = (const float*)d_in[6];
    const float* b1   = (const float*)d_in[7];
    const float* W2   = (const float*)d_in[8];
    const float* b2   = (const float*)d_in[9];
    const float* ln1w = (const float*)d_in[10];
    const float* ln1b = (const float*)d_in[11];
    const float* ln2w = (const float*)d_in[12];
    const float* ln2b = (const float*)d_in[13];

    const int Bsz = 8, T = 1024, D = 768, FF = 3072, H = 12;
    const int M = Bsz * T;  // 8192

    char* ws = (char*)d_ws;
    size_t off = 0;
    auto alloc = [&](size_t bytes) -> void* {
        void* p = ws + off;
        off += (bytes + 255) & ~(size_t)255;
        return p;
    };
    unsigned short* wqb = (unsigned short*)alloc((size_t)D * D * 2);
    unsigned short* wkb = (unsigned short*)alloc((size_t)D * D * 2);
    unsigned short* wvb = (unsigned short*)alloc((size_t)D * D * 2);
    unsigned short* wob = (unsigned short*)alloc((size_t)D * D * 2);
    unsigned short* w1b = (unsigned short*)alloc((size_t)FF * D * 2);
    unsigned short* w2b = (unsigned short*)alloc((size_t)D * FF * 2);
    unsigned short* xn  = (unsigned short*)alloc((size_t)M * D * 2);
    unsigned short* qb  = (unsigned short*)alloc((size_t)M * D * 2);
    unsigned short* kb  = (unsigned short*)alloc((size_t)M * D * 2);
    unsigned short* vb  = (unsigned short*)alloc((size_t)M * D * 2);
    unsigned short* ob  = (unsigned short*)alloc((size_t)M * D * 2);
    float*          hb  = (float*)alloc((size_t)M * D * 4);
    unsigned short* zb  = (unsigned short*)alloc((size_t)M * D * 2);
    unsigned short* act = (unsigned short*)alloc((size_t)M * FF * 2);

    int ndd = D * D, ndf = FF * D;
    cvt_bf16_kernel<<<(ndd + 255) / 256, 256, 0, stream>>>(Wq, wqb, ndd);
    cvt_bf16_kernel<<<(ndd + 255) / 256, 256, 0, stream>>>(Wk, wkb, ndd);
    cvt_bf16_kernel<<<(ndd + 255) / 256, 256, 0, stream>>>(Wv, wvb, ndd);
    cvt_bf16_kernel<<<(ndd + 255) / 256, 256, 0, stream>>>(Wo, wob, ndd);
    cvt_bf16_kernel<<<(ndf + 255) / 256, 256, 0, stream>>>(W1, w1b, ndf);
    cvt_bf16_kernel<<<(ndf + 255) / 256, 256, 0, stream>>>(W2, w2b, ndf);

    layernorm_bf16_kernel<<<M, 256, 0, stream>>>(x, ln1w, ln1b, xn, D);

    dim3 g1(D / 128, M / 128);    // (6, 64)
    gemm_nt_bf16_kernel<0><<<g1, 256, 0, stream>>>(xn, wqb, M, D, D, qb, nullptr, nullptr);
    gemm_nt_bf16_kernel<0><<<g1, 256, 0, stream>>>(xn, wkb, M, D, D, kb, nullptr, nullptr);
    gemm_nt_bf16_kernel<0><<<g1, 256, 0, stream>>>(xn, wvb, M, D, D, vb, nullptr, nullptr);

    dim3 ga(T / 64, H, Bsz);      // (16, 12, 8)
    attn_kernel<<<ga, 128, 0, stream>>>(qb, kb, vb, ob);

    gemm_nt_bf16_kernel<1><<<g1, 256, 0, stream>>>(ob, wob, M, D, D, hb, bo, x);

    layernorm_bf16_kernel<<<M, 256, 0, stream>>>(hb, ln2w, ln2b, zb, D);

    dim3 g2(FF / 128, M / 128);   // (24, 64)
    gemm_nt_bf16_kernel<2><<<g2, 256, 0, stream>>>(zb, w1b, M, FF, D, act, b1, nullptr);

    gemm_nt_bf16_kernel<1><<<g1, 256, 0, stream>>>(act, w2b, M, D, FF, (float*)d_out, b2, hb);
}
